// GTNLayer_90228672955101
// MI455X (gfx1250) — matrix operand order repaired and checked by reading, not compiled, not run on hardware
//
#include <hip/hip_runtime.h>
#include <math.h>

#define NN 50000
#define KK 32
#define DD 128
#define RSQRT_D 0.08838834764831845f   // 1/sqrt(128)

typedef __attribute__((ext_vector_type(16))) __bf16 v16bf;
typedef __attribute__((ext_vector_type(8)))  float  v8f;

union BFfrag { v16bf v; unsigned int u[8]; };
union BFpk   { __bf16 h[2]; unsigned int u; };

__device__ __forceinline__ v8f wmma_bf16(v16bf a, v16bf b, v8f c) {
    // D = A(16x32 bf16) * B(32x16 bf16) + C(16x16 f32)
    return __builtin_amdgcn_wmma_f32_16x16x32_bf16(
        false, a, false, b, (short)0, c, false, false);
}

// A-fragment (16x32 bf16) from LDS tile, row pitch `pitch`, K-offset kk0.
// Layout (ISA 7.12.2): lanes 0-15 row M=lane, K 0-7 & 16-23; lanes 16-31 row M=lane-16, K 8-15 & 24-31.
__device__ __forceinline__ v16bf load_a_frag(const __bf16* tile, int pitch,
                                             int kk0, int lane) {
    int r  = lane & 15;
    int kb = (lane >> 4) * 8;
    const unsigned int* p0 = (const unsigned int*)(tile + r * pitch + kk0 + kb);
    const unsigned int* p1 = (const unsigned int*)(tile + r * pitch + kk0 + kb + 16);
    BFfrag f;
#pragma unroll
    for (int i = 0; i < 4; i++) { f.u[i] = p0[i]; f.u[4 + i] = p1[i]; }
    return f.v;
}

// B-fragment (32x16 bf16). tileT stored [col][kk] with row pitch `pitch`.
// lanes 0-15: column lane, K 0-7 & 16-23; lanes 16-31: column lane-16, K 8-15 & 24-31.
__device__ __forceinline__ v16bf load_b_frag(const __bf16* tileT, int pitch,
                                             int col0, int kk0, int lane) {
    int n  = lane & 15;
    int kb = (lane >> 4) * 8;
    const unsigned int* p = (const unsigned int*)(tileT + (col0 + n) * pitch + kk0 + kb);
    BFfrag f;
#pragma unroll
    for (int i = 0; i < 4; i++) { f.u[i] = p[i]; f.u[4 + i] = p[8 + i]; }
    return f.v;
}

// A-fragment built directly from global fp32 rows (hardware f32->bf16 convert).
__device__ __forceinline__ v16bf load_a_frag_global_f32(const float* base, int rowStride,
                                                        int kk0, int lane, int nb) {
    int r    = lane & 15;
    int node = nb + r; if (node >= NN) node = NN - 1;   // clamp: store side is guarded
    int kb   = (lane >> 4) * 8;
    const float* p = base + (size_t)node * rowStride + kk0 + kb;
    BFfrag f;
#pragma unroll
    for (int i = 0; i < 4; i++) {
        BFpk a, b;
        a.h[0] = (__bf16)p[2 * i];      a.h[1] = (__bf16)p[2 * i + 1];
        b.h[0] = (__bf16)p[16 + 2 * i]; b.h[1] = (__bf16)p[16 + 2 * i + 1];
        f.u[i] = a.u; f.u[4 + i] = b.u;
    }
    return f.v;
}

// ---------------------------------------------------------------------------
// Kernel 1: key = tgt@k_w + k_b ; w = key@q_w^T ; c = key.q_b ;
//           s[n,k] = (neigh[n,k].w[n] + c[n]) / sqrt(D)
// Block = 64 threads (2 waves), 16 nodes per wave, 32 nodes per block.
// ---------------------------------------------------------------------------
__global__ void __launch_bounds__(64) k1_scores(
    const float* __restrict__ tgt, const float* __restrict__ neigh,
    const float* __restrict__ kw,  const float* __restrict__ kb,
    const float* __restrict__ qw,  const float* __restrict__ qb,
    float* __restrict__ sOut)
{
    __shared__ __bf16 ldsW[DD * DD];              // 32 KB: k_w^T, then q_w (reused)
    __shared__ float ldsKB[DD];
    __shared__ float ldsQB[DD];
    __shared__ __bf16 keyB[2][16 * DD];           // 8 KB : key tile, bf16 [m][e]
    __shared__ float  wT[2][16 * DD];             // 16 KB: w tile, f32 [m][d]
    __shared__ float  cT[2][16];

    const int tid = threadIdx.x, wv = tid >> 5, lane = tid & 31;
    const int nb  = blockIdx.x * 32 + wv * 16;

    // stage k_w transposed: ldsW[e*DD+d] = k_w[d][e]   (coalesced global reads)
    for (int i = tid; i < DD * DD; i += 64) {
        int d = i >> 7, e = i & 127;
        ldsW[e * DD + d] = (__bf16)kw[i];
    }
    if (tid < DD) { ldsKB[tid] = kb[tid]; ldsQB[tid] = qb[tid]; }
    __syncthreads();

    // phase A: key = tgt @ k_w + k_b
    v8f acc[8];
#pragma unroll
    for (int t = 0; t < 8; t++)
#pragma unroll
        for (int j = 0; j < 8; j++) acc[t][j] = 0.f;

#pragma unroll
    for (int ks = 0; ks < 4; ks++) {
        v16bf a = load_a_frag_global_f32(tgt, DD, ks * 32, lane, nb);
#pragma unroll
        for (int t = 0; t < 8; t++) {
            v16bf b = load_b_frag(ldsW, DD, t * 16, ks * 32, lane);
            acc[t] = wmma_bf16(a, b, acc[t]);
        }
    }
    {   // C layout: vgpr j -> row j (lanes 0-15) / j+8 (lanes 16-31), col = lane%16
        int n16 = lane & 15, mo = (lane >> 4) * 8;
#pragma unroll
        for (int t = 0; t < 8; t++) {
            int e = t * 16 + n16;
            float bias = ldsKB[e];
#pragma unroll
            for (int j = 0; j < 8; j++)
                keyB[wv][(j + mo) * DD + e] = (__bf16)(acc[t][j] + bias);
        }
    }
    __syncthreads();

    // restage ldsW = q_w row-major; c[m] = key[m].q_b
    for (int i = tid; i < DD * DD; i += 64) ldsW[i] = (__bf16)qw[i];
    if (lane < 16) {
        float c = 0.f;
#pragma unroll 8
        for (int e = 0; e < DD; e++) c += (float)keyB[wv][lane * DD + e] * ldsQB[e];
        cT[wv][lane] = c;
    }
    __syncthreads();

    // phase B: w = key @ q_w^T   (B[kk=e][nn=d] = q_w[d][e] -> row-major q_w is [col][kk])
#pragma unroll
    for (int t = 0; t < 8; t++)
#pragma unroll
        for (int j = 0; j < 8; j++) acc[t][j] = 0.f;
#pragma unroll
    for (int ks = 0; ks < 4; ks++) {
        v16bf a = load_a_frag(keyB[wv], DD, ks * 32, lane);
#pragma unroll
        for (int t = 0; t < 8; t++) {
            v16bf b = load_b_frag(ldsW, DD, t * 16, ks * 32, lane);
            acc[t] = wmma_bf16(a, b, acc[t]);
        }
    }
    {
        int n16 = lane & 15, mo = (lane >> 4) * 8;
#pragma unroll
        for (int t = 0; t < 8; t++) {
            int d = t * 16 + n16;
#pragma unroll
            for (int j = 0; j < 8; j++) wT[wv][(j + mo) * DD + d] = acc[t][j];
        }
    }
    __syncthreads();

    // phase C: scores. lane = neighbor index k, loop over this wave's 16 nodes.
    for (int m = 0; m < 16; m++) {
        int node = nb + m;
        if (node >= NN) break;                       // wave-uniform
        // prefetch next node's row for this lane
        int nodeP = (node + 1 < NN) ? node + 1 : node;
        __builtin_prefetch(neigh + ((size_t)nodeP * KK + lane) * DD, 0, 1);
        const float4* nrow = (const float4*)(neigh + ((size_t)node * KK + lane) * DD);
        const float4* wrow = (const float4*)(wT[wv] + m * DD);
        float s = 0.f;
#pragma unroll
        for (int d4 = 0; d4 < DD / 4; d4++) {
            float4 nv = nrow[d4], wvv = wrow[d4];
            s += nv.x * wvv.x + nv.y * wvv.y + nv.z * wvv.z + nv.w * wvv.w;
        }
        sOut[(size_t)node * KK + lane] = (s + cT[wv][m]) * RSQRT_D;
    }
}

// ---------------------------------------------------------------------------
// Kernel 2: softmax stats over the NODE axis, one block per neighbor column.
// ---------------------------------------------------------------------------
__global__ void __launch_bounds__(256) k2_stats(const float* __restrict__ s,
                                                float* __restrict__ stats)
{
    __shared__ float red[256];
    const int col = blockIdx.x, t = threadIdx.x;
    float m = -3.4e38f;
    for (int n = t; n < NN; n += 256) m = fmaxf(m, s[(size_t)n * KK + col]);
    red[t] = m; __syncthreads();
    for (int off = 128; off > 0; off >>= 1) {
        if (t < off) red[t] = fmaxf(red[t], red[t + off]);
        __syncthreads();
    }
    const float mx = red[0]; __syncthreads();
    float sum = 0.f;
    for (int n = t; n < NN; n += 256) sum += __expf(s[(size_t)n * KK + col] - mx);
    red[t] = sum; __syncthreads();
    for (int off = 128; off > 0; off >>= 1) {
        if (t < off) red[t] += red[t + off];
        __syncthreads();
    }
    if (t == 0) { stats[col] = mx; stats[KK + col] = red[0]; }
}

// ---------------------------------------------------------------------------
// Kernel 3: att = exp(s-max)/sum ; a[n] = sum_k att*neigh[n,k] ;
//           h[n] = a[n]@v_w + (sum_k att)*v_b
// Block = 128 threads (4 waves), 16 nodes per wave, 64 nodes per block.
// ---------------------------------------------------------------------------
__global__ void __launch_bounds__(128) k3_output(
    const float* __restrict__ neigh, const float* __restrict__ sArr,
    const float* __restrict__ stats, const float* __restrict__ vw,
    const float* __restrict__ vb,    float* __restrict__ out)
{
    __shared__ union LdsU {                       // 32 KB, two lifetimes:
        __bf16 w[DD * DD];                        //  - v_w^T [e][d] during WMMA
        float  o[4][16 * DD];                     //  - per-wave f32 out tiles after
    } ldsU;
    __shared__ float ldsVB[DD];
    __shared__ float ldsMax[KK];
    __shared__ float ldsSum[KK];
    __shared__ float attL[4][16 * KK];            // 8 KB
    __shared__ __bf16 aT[4][16 * DD];             // 16 KB, bf16 [m][d]
    __shared__ float sAtt[4][16];

    const int tid = threadIdx.x, wv = tid >> 5, lane = tid & 31;
    const int nb  = blockIdx.x * 64 + wv * 16;

    for (int i = tid; i < DD * DD; i += 128) {
        int d = i >> 7, e = i & 127;
        ldsU.w[e * DD + d] = (__bf16)vw[i];       // B[kk=d][nn=e] = v_w[d][e]
    }
    if (tid < DD) ldsVB[tid] = vb[tid];
    if (tid < KK) { ldsMax[tid] = stats[tid]; ldsSum[tid] = stats[KK + tid]; }
    __syncthreads();

    // attention tile: lane = k
    for (int m = 0; m < 16; m++) {
        int node = nb + m; if (node >= NN) node = NN - 1;
        float sv = sArr[(size_t)node * KK + lane];
        attL[wv][m * KK + lane] = __expf(sv - ldsMax[lane]) / ldsSum[lane];
    }
    __syncthreads();
    if (lane < 16) {
        float s = 0.f;
#pragma unroll
        for (int k = 0; k < KK; k++) s += attL[wv][lane * KK + k];
        sAtt[wv][lane] = s;
    }
    // a[m][d] = sum_k att[m,k]*neigh[m,k,d]; lane covers 4 dims (fully coalesced rows)
    for (int m = 0; m < 16; m++) {
        int node = nb + m; int nodeL = node < NN ? node : NN - 1;
        int nodeP = (node + 1 < NN) ? node + 1 : nodeL;
        __builtin_prefetch((const float4*)(neigh + (size_t)nodeP * KK * DD) + lane, 0, 1);
        const float4* base = (const float4*)(neigh + (size_t)nodeL * KK * DD) + lane;
        float4 a4 = {0.f, 0.f, 0.f, 0.f};
#pragma unroll 4
        for (int k = 0; k < KK; k++) {
            float av = attL[wv][m * KK + k];
            float4 nv = base[(size_t)k * (DD / 4)];
            a4.x += av * nv.x; a4.y += av * nv.y; a4.z += av * nv.z; a4.w += av * nv.w;
        }
        __bf16* dst = aT[wv] + m * DD + lane * 4;
        dst[0] = (__bf16)a4.x; dst[1] = (__bf16)a4.y;
        dst[2] = (__bf16)a4.z; dst[3] = (__bf16)a4.w;
    }
    __syncthreads();

    // h tile = aT @ v_w  (WMMA)
    v8f acc[8];
#pragma unroll
    for (int t = 0; t < 8; t++)
#pragma unroll
        for (int j = 0; j < 8; j++) acc[t][j] = 0.f;
#pragma unroll
    for (int ks = 0; ks < 4; ks++) {
        v16bf a = load_a_frag(aT[wv], DD, ks * 32, lane);
#pragma unroll
        for (int t = 0; t < 8; t++) {
            v16bf b = load_b_frag(ldsU.w, DD, t * 16, ks * 32, lane);
            acc[t] = wmma_bf16(a, b, acc[t]);
        }
    }
    __syncthreads();                              // ldsU.w dead -> reuse as ldsU.o

    // epilogue: C tiles -> LDS f32, then fully coalesced float4 row stores
    {
        float* myOut = ldsU.o[wv];
        const int n16 = lane & 15, mo = (lane >> 4) * 8;
#pragma unroll
        for (int t = 0; t < 8; t++) {
            int e = t * 16 + n16;
            float bias = ldsVB[e];
#pragma unroll
            for (int j = 0; j < 8; j++)
                myOut[(mo + j) * DD + e] = acc[t][j] + sAtt[wv][mo + j] * bias;
        }
    }
    __syncthreads();
    {
        const float* myOut = ldsU.o[wv];
        for (int r = 0; r < 16; r++) {
            int node = nb + r;
            if (node >= NN) break;                // wave-uniform
            float4 v4 = ((const float4*)(myOut + r * DD))[lane];
            ((float4*)(out + (size_t)node * DD))[lane] = v4;
        }
    }
}

extern "C" void kernel_launch(void* const* d_in, const int* in_sizes, int n_in,
                              void* d_out, int out_size, void* d_ws, size_t ws_size,
                              hipStream_t stream)
{
    const float* tgt   = (const float*)d_in[0];
    const float* neigh = (const float*)d_in[1];
    const float* kw    = (const float*)d_in[2];
    const float* kb    = (const float*)d_in[3];
    const float* qw    = (const float*)d_in[4];
    const float* qb    = (const float*)d_in[5];
    const float* vw    = (const float*)d_in[6];
    const float* vb    = (const float*)d_in[7];
    float* out = (float*)d_out;

    float* sArr  = (float*)d_ws;                 // N*32 scores  (6.4 MB)
    float* stats = sArr + (size_t)NN * KK;       // 32 max + 32 sum

    const int nblk1 = (NN + 31) / 32;            // 32 nodes / block (2 waves)
    const int nblk3 = (NN + 63) / 64;            // 64 nodes / block (4 waves)

    k1_scores<<<nblk1, 64, 0, stream>>>(tgt, neigh, kw, kb, qw, qb, sArr);
    k2_stats <<<KK,   256, 0, stream>>>(sArr, stats);
    k3_output<<<nblk3, 128, 0, stream>>>(neigh, sArr, stats, vw, vb, out);
}
